// MaskedAttention_60387240182436
// MI455X (gfx1250) — matrix operand order repaired and checked by reading, not compiled, not run on hardware
//
#include <hip/hip_runtime.h>
#include <hip/hip_bf16.h>

// ---------------------------------------------------------------------------
// MaskedAttention for MI455X (gfx1250, wave32, WMMA)
//   x[B,N,D] --LN--> xn(f16) --WMMA GEMM--> q,k,v (f16, per-head; v transposed)
//   flash-style attention per 16-row strip (scores never hit HBM)
//   out projection WMMA GEMM -> f32 output
// GEMMs stage the shared weight tile in LDS via async global->LDS DMA
// (ASYNCcnt path), double-buffered so the fetch overlaps the WMMAs.
// ---------------------------------------------------------------------------

typedef __attribute__((ext_vector_type(16))) _Float16 v16h;
typedef __attribute__((ext_vector_type(8)))  _Float16 v8h;
typedef __attribute__((ext_vector_type(8)))  float    v8f;
typedef __attribute__((ext_vector_type(4)))  int      v4i;

typedef __attribute__((address_space(1))) v4i* gv4p;   // global 128-bit payload
typedef __attribute__((address_space(3))) v4i* lv4p;   // LDS 128-bit payload

#define B_     2
#define N_     2048
#define D_     1024
#define HEADS  16
#define DHEAD  64
#define INNER  1024            // HEADS*DHEAD
#define SCALE  0.125f          // 64^-0.5
#define LN_EPS 1e-5f

__device__ __forceinline__ v8f wmma_f16(v16h a, v16h b, v8f c) {
  return __builtin_amdgcn_wmma_f32_16x16x32_f16(
      /*neg_a=*/false, a, /*neg_b=*/false, b,
      /*c_mod=*/(short)0, c, /*reuse_a=*/false, /*reuse_b=*/false);
}

// A-fragment (16x32 f16): lane holds row (lane&15); contiguous K = (lane>>4)*16
// .. +15 relative to kbase.  Two 16B loads.
__device__ __forceinline__ v16h load_a_frag(const _Float16* rowptr, int kbase, int lane) {
  const int c0 = kbase + ((lane >> 4) << 4);
  v8h lo = *(const v8h*)(rowptr + c0);
  v8h hi = *(const v8h*)(rowptr + c0 + 8);
  v16h a;
#pragma unroll
  for (int i = 0; i < 8; ++i) { a[i] = lo[i]; a[i + 8] = hi[i]; }
  return a;
}

// ---------------------------------------------------------------------------
// LayerNorm: one block per row, 256 threads, f32 stats, f16 output
// ---------------------------------------------------------------------------
__global__ __launch_bounds__(256) void ln_kernel(const float* __restrict__ x,
                                                 const float* __restrict__ gamma,
                                                 const float* __restrict__ beta,
                                                 _Float16* __restrict__ xn) {
  __shared__ float red[16];
  const int row = blockIdx.x;
  const float* xr = x + (size_t)row * D_;
  float vals[4];
  float s = 0.f, ss = 0.f;
#pragma unroll
  for (int i = 0; i < 4; ++i) {
    float v = xr[threadIdx.x + i * 256];
    vals[i] = v; s += v; ss += v * v;
  }
#pragma unroll
  for (int msk = 1; msk < 32; msk <<= 1) {
    s  += __shfl_xor(s,  msk);
    ss += __shfl_xor(ss, msk);
  }
  const int wave = threadIdx.x >> 5, lane = threadIdx.x & 31;
  if (lane == 0) { red[wave] = s; red[8 + wave] = ss; }
  __syncthreads();
  if (threadIdx.x == 0) {
    float ts = 0.f, tss = 0.f;
#pragma unroll
    for (int w = 0; w < 8; ++w) { ts += red[w]; tss += red[8 + w]; }
    red[0] = ts; red[8] = tss;
  }
  __syncthreads();
  const float mu  = red[0] * (1.0f / D_);
  const float var = red[8] * (1.0f / D_) - mu * mu;
  const float inv = rsqrtf(var + LN_EPS);
  _Float16* xo = xn + (size_t)row * D_;
#pragma unroll
  for (int i = 0; i < 4; ++i) {
    const int c = threadIdx.x + i * 256;
    xo[c] = (_Float16)((vals[i] - mu) * inv * gamma[c] + beta[c]);
  }
}

// ---------------------------------------------------------------------------
// Transpose + f32->f16: dst[n*K + k] = src[k*Nn + n]   (dst is [Nn][K])
// ---------------------------------------------------------------------------
__global__ void tconv_kernel(const float* __restrict__ src, _Float16* __restrict__ dst,
                             int K, int Nn) {
  const size_t idx = (size_t)blockIdx.x * blockDim.x + threadIdx.x;
  if (idx >= (size_t)K * Nn) return;
  const int k = (int)(idx % K);
  const int n = (int)(idx / K);
  dst[idx] = (_Float16)src[(size_t)k * Nn + n];
}

// ---------------------------------------------------------------------------
// Shared-LDS weight tile helpers for the projection GEMMs.
// Tile: 64 output cols x 32 K, stored with row stride 40 halves (80B) to
// spread LDS banks.  Fetched with async global->LDS b128 (one per thread).
// ---------------------------------------------------------------------------
#define BROWS 64
#define BSTRIDE 40

__device__ __forceinline__ void async_fetch_btile(const _Float16* gsrc,
                                                  _Float16 (*buf)[BSTRIDE],
                                                  int lrow, int lchunk) {
  _Float16* g = const_cast<_Float16*>(gsrc);
  __builtin_amdgcn_global_load_async_to_lds_b128(
      (gv4p)g, (lv4p)&buf[lrow][lchunk], /*offset=*/0, /*cpol=*/0);
}

__device__ __forceinline__ v16h lds_b_frag(const _Float16 (*buf)[BSTRIDE],
                                           int t, int col, int halfu) {
  const _Float16* bp = &buf[t * 16 + col][halfu << 4];
  v8h b0 = *(const v8h*)bp;
  v8h b1 = *(const v8h*)(bp + 8);
  v16h b;
#pragma unroll
  for (int i = 0; i < 8; ++i) { b[i] = b0[i]; b[i + 8] = b1[i]; }
  return b;
}

// ---------------------------------------------------------------------------
// QKV GEMM: C[4096,3072] = xn[4096,1024] @ Wqkv ; wt = Wqkv^T [3072,1024] f16.
// Block = 128 rows x 64 cols (8 waves x 16 rows share one LDS weight tile,
// double-buffered via async DMA).  Writes q,k row-major per head and v
// transposed [B,H,DHEAD,N] for the attention B-operand.
// ---------------------------------------------------------------------------
__global__ __launch_bounds__(256) void qkv_gemm_kernel(const _Float16* __restrict__ xn,
                                                       const _Float16* __restrict__ wt,
                                                       _Float16* __restrict__ qh,
                                                       _Float16* __restrict__ kh,
                                                       _Float16* __restrict__ vt) {
  __shared__ _Float16 bbuf[2][BROWS][BSTRIDE];   // 10 KB
  const int lane = threadIdx.x & 31;
  const int wave = threadIdx.x >> 5;
  const int ntiles = (3 * INNER) / 64;           // 48 col groups
  const int i0 = (blockIdx.x / ntiles) * 128 + wave * 16;
  const int n0 = (blockIdx.x % ntiles) * 64;
  const int col = lane & 15, halfu = lane >> 4;

  const int lrow   = threadIdx.x >> 2;           // 0..63
  const int lchunk = (threadIdx.x & 3) << 3;     // 0,8,16,24 halves
  const _Float16* gsrc = wt + (size_t)(n0 + lrow) * D_ + lchunk;

  async_fetch_btile(gsrc, bbuf[0], lrow, lchunk);        // prologue: tile 0

  v8f acc[4];
#pragma unroll
  for (int t = 0; t < 4; ++t) acc[t] = 0.f;

  const _Float16* arow = xn + (size_t)(i0 + col) * D_;
  int cur = 0;
  for (int kk = 0; kk < D_; kk += 32, cur ^= 1) {
    asm volatile("s_wait_asynccnt 0" ::: "memory");
    __syncthreads();                                     // tile[cur] ready for all
    if (kk + 32 < D_)                                    // prefetch next tile
      async_fetch_btile(gsrc + kk + 32, bbuf[cur ^ 1], lrow, lchunk);
    const v16h a = load_a_frag(arow, kk, lane);
#pragma unroll
    for (int t = 0; t < 4; ++t)
      acc[t] = wmma_f16(a, lds_b_frag(bbuf[cur], t, col, halfu), acc[t]);
    __syncthreads();                                     // done reading tile[cur]
  }

  const int section = n0 / INNER;                // 0=q 1=k 2=v
  const int h = (n0 % INNER) / DHEAD;
#pragma unroll
  for (int t = 0; t < 4; ++t) {
    const int d = t * 16 + col;                  // n0 is 64-aligned
#pragma unroll
    for (int r = 0; r < 8; ++r) {
      const int row = i0 + r + 8 * halfu;        // global token row
      const int b = row / N_, tok = row % N_;
      const size_t bh = (size_t)(b * HEADS + h);
      const _Float16 val = (_Float16)acc[t][r];
      if (section == 0)      qh[(bh * N_ + tok) * DHEAD + d] = val;
      else if (section == 1) kh[(bh * N_ + tok) * DHEAD + d] = val;
      else                   vt[(bh * DHEAD + d) * N_ + tok] = val;
    }
  }
}

// ---------------------------------------------------------------------------
// Flash attention: one wave owns 16 query rows of one (b,h); streams K/V in
// 32-column steps.  Mask pre-loaded into the WMMA C operand so D = qk + m,
// then scaled (matches reference order).  P goes through LDS to convert the
// C/D layout into an A-fragment for the P*V WMMA.
// ---------------------------------------------------------------------------
__global__ __launch_bounds__(256) void attn_kernel(const _Float16* __restrict__ qh,
                                                   const _Float16* __restrict__ kh,
                                                   const _Float16* __restrict__ vt,
                                                   const float* __restrict__ mask,
                                                   _Float16* __restrict__ ao) {
  __shared__ _Float16 ldsP[8][16][32];           // per-wave 16x32 P tile (8KB)
  const int lane = threadIdx.x & 31;
  const int wave = threadIdx.x >> 5;
  const int strip = blockIdx.x * 8 + wave;       // [0, B*H*N/16)
  const int b   = strip / (HEADS * (N_ / 16));
  const int rem = strip % (HEADS * (N_ / 16));
  const int h   = rem / (N_ / 16);
  const int i0  = (rem % (N_ / 16)) * 16;
  const int col = lane & 15, halfu = lane >> 4;
  const size_t bh = (size_t)(b * HEADS + h);

  // Q fragments for K-dim 0..31 and 32..63, kept in registers for the loop.
  const _Float16* qrow = qh + (bh * N_ + i0 + col) * DHEAD;
  const v16h qa0 = load_a_frag(qrow, 0, lane);
  const v16h qa1 = load_a_frag(qrow, 32, lane);

  v8f o[4];
#pragma unroll
  for (int t = 0; t < 4; ++t) o[t] = 0.f;
  float mstat[8], lstat[8];
#pragma unroll
  for (int r = 0; r < 8; ++r) { mstat[r] = -1e30f; lstat[r] = 0.f; }

  const float* mbase = mask + ((size_t)b * N_ + i0) * N_;

  for (int j = 0; j < N_; j += 32) {
    // ---- scores: two 16x16 tiles, C seeded with mask (pre-scale) ----
    v8f c0, c1;
#pragma unroll
    for (int r = 0; r < 8; ++r) {
      const float* mr = mbase + (size_t)(r + 8 * halfu) * N_;
      c0[r] = mr[j + col];
      c1[r] = mr[j + 16 + col];
    }
    const _Float16* k0 = kh + (bh * N_ + j + col) * DHEAD + (halfu << 4);
    c0 = wmma_f16(qa0, *(const v16h*)(k0), c0);
    c0 = wmma_f16(qa1, *(const v16h*)(k0 + 32), c0);
    const _Float16* k1 = k0 + (size_t)16 * DHEAD;
    c1 = wmma_f16(qa0, *(const v16h*)(k1), c1);
    c1 = wmma_f16(qa1, *(const v16h*)(k1 + 32), c1);

    // ---- streaming softmax (per-row stats replicated in each 16-lane half) ----
    float alpha[8];
#pragma unroll
    for (int r = 0; r < 8; ++r) {
      const float s0 = c0[r] * SCALE;
      const float s1 = c1[r] * SCALE;
      float mx = fmaxf(s0, s1);
      mx = fmaxf(mx, __shfl_xor(mx, 1));
      mx = fmaxf(mx, __shfl_xor(mx, 2));
      mx = fmaxf(mx, __shfl_xor(mx, 4));
      mx = fmaxf(mx, __shfl_xor(mx, 8));
      const float mnew = fmaxf(mstat[r], mx);
      const float a  = __expf(mstat[r] - mnew);
      const float p0 = __expf(s0 - mnew);
      const float p1 = __expf(s1 - mnew);
      float ps = p0 + p1;
      ps += __shfl_xor(ps, 1);
      ps += __shfl_xor(ps, 2);
      ps += __shfl_xor(ps, 4);
      ps += __shfl_xor(ps, 8);
      lstat[r] = lstat[r] * a + ps;
      mstat[r] = mnew;
      alpha[r] = a;
      ldsP[wave][r + 8 * halfu][col]      = (_Float16)p0;
      ldsP[wave][r + 8 * halfu][16 + col] = (_Float16)p1;
    }

    // D-layout -> A-layout via LDS; DS ops are wave-private, wait on dscnt.
    asm volatile("s_wait_dscnt 0" ::: "memory");
    v16h pa;
    {
      const int pr  = col;
      const int pc0 = halfu << 4;
#pragma unroll
      for (int i = 0; i < 8; ++i) {
        pa[i]     = ldsP[wave][pr][pc0 + i];
        pa[i + 8] = ldsP[wave][pr][pc0 + 8 + i];
      }
    }

    // ---- O = diag(alpha)*O + P @ V  (V pre-transposed: [B,H,DHEAD,N]) ----
#pragma unroll
    for (int t = 0; t < 4; ++t) {
#pragma unroll
      for (int r = 0; r < 8; ++r) o[t][r] *= alpha[r];
      const _Float16* vp = vt + (bh * DHEAD + t * 16 + col) * N_ + j + (halfu << 4);
      o[t] = wmma_f16(pa, *(const v16h*)vp, o[t]);
    }
  }

  // ---- finalize and write f16 for the output projection ----
  float rcp[8];
#pragma unroll
  for (int r = 0; r < 8; ++r) rcp[r] = 1.0f / lstat[r];
#pragma unroll
  for (int t = 0; t < 4; ++t)
#pragma unroll
    for (int r = 0; r < 8; ++r) {
      const int tok = i0 + r + 8 * halfu;
      ao[((size_t)(b * N_ + tok)) * INNER + h * DHEAD + t * 16 + col] =
          (_Float16)(o[t][r] * rcp[r]);
    }
}

// ---------------------------------------------------------------------------
// Output GEMM: out[4096,1024] = ao[4096,1024] @ W_out ; wt = W_out^T, f32 out.
// Same block structure as qkv_gemm: LDS weight tile + async double buffering.
// ---------------------------------------------------------------------------
__global__ __launch_bounds__(256) void out_gemm_kernel(const _Float16* __restrict__ ao,
                                                       const _Float16* __restrict__ wt,
                                                       float* __restrict__ out) {
  __shared__ _Float16 bbuf[2][BROWS][BSTRIDE];   // 10 KB
  const int lane = threadIdx.x & 31;
  const int wave = threadIdx.x >> 5;
  const int ntiles = D_ / 64;                    // 16 col groups
  const int i0 = (blockIdx.x / ntiles) * 128 + wave * 16;
  const int n0 = (blockIdx.x % ntiles) * 64;
  const int col = lane & 15, halfu = lane >> 4;

  const int lrow   = threadIdx.x >> 2;
  const int lchunk = (threadIdx.x & 3) << 3;
  const _Float16* gsrc = wt + (size_t)(n0 + lrow) * INNER + lchunk;

  async_fetch_btile(gsrc, bbuf[0], lrow, lchunk);

  v8f acc[4];
#pragma unroll
  for (int t = 0; t < 4; ++t) acc[t] = 0.f;

  const _Float16* arow = ao + (size_t)(i0 + col) * INNER;
  int cur = 0;
  for (int kk = 0; kk < INNER; kk += 32, cur ^= 1) {
    asm volatile("s_wait_asynccnt 0" ::: "memory");
    __syncthreads();
    if (kk + 32 < INNER)
      async_fetch_btile(gsrc + kk + 32, bbuf[cur ^ 1], lrow, lchunk);
    const v16h a = load_a_frag(arow, kk, lane);
#pragma unroll
    for (int t = 0; t < 4; ++t)
      acc[t] = wmma_f16(a, lds_b_frag(bbuf[cur], t, col, halfu), acc[t]);
    __syncthreads();
  }
#pragma unroll
  for (int t = 0; t < 4; ++t)
#pragma unroll
    for (int r = 0; r < 8; ++r)
      out[(size_t)(i0 + r + 8 * halfu) * D_ + n0 + t * 16 + col] = acc[t][r];
}

// ---------------------------------------------------------------------------
// Workspace layout (bytes):
//   xn_h    [4096,1024] f16 : 8 MiB    @ 0
//   wqkv_t  [3072,1024] f16 : 6 MiB    @ 8 MiB
//   wout_t  [1024,1024] f16 : 2 MiB    @ 14 MiB
//   q_h  [B,H,N,64]     f16 : 8 MiB    @ 16 MiB
//   k_h  [B,H,N,64]     f16 : 8 MiB    @ 24 MiB
//   v_t  [B,H,64,N]     f16 : 8 MiB    @ 32 MiB
//   ao_h [4096,1024]    f16 : 8 MiB    @ 40 MiB   (total 48 MiB)
// ---------------------------------------------------------------------------
extern "C" void kernel_launch(void* const* d_in, const int* in_sizes, int n_in,
                              void* d_out, int out_size, void* d_ws, size_t ws_size,
                              hipStream_t stream) {
  (void)in_sizes; (void)n_in; (void)out_size;
  const float* x     = (const float*)d_in[0];
  const float* m     = (const float*)d_in[1];
  const float* gamma = (const float*)d_in[2];
  const float* beta  = (const float*)d_in[3];
  const float* Wqkv  = (const float*)d_in[4];
  const float* Wout  = (const float*)d_in[5];
  float* out = (float*)d_out;

  char* ws = (char*)d_ws;
  const size_t MB = 1024ull * 1024ull;
  _Float16* xn_h   = (_Float16*)(ws + 0 * MB);
  _Float16* wqkv_t = (_Float16*)(ws + 8 * MB);
  _Float16* wout_t = (_Float16*)(ws + 14 * MB);
  _Float16* q_h    = (_Float16*)(ws + 16 * MB);
  _Float16* k_h    = (_Float16*)(ws + 24 * MB);
  _Float16* v_t    = (_Float16*)(ws + 32 * MB);
  _Float16* ao_h   = (_Float16*)(ws + 40 * MB);
  if (ws_size < 48 * MB) return;  // workspace contract: need 48 MiB

  // 1) LayerNorm -> f16
  ln_kernel<<<B_ * N_, 256, 0, stream>>>(x, gamma, beta, xn_h);
  // 2) transpose+convert weights
  tconv_kernel<<<(3 * INNER * D_) / 256, 256, 0, stream>>>(Wqkv, wqkv_t, D_, 3 * INNER);
  tconv_kernel<<<(INNER * D_) / 256, 256, 0, stream>>>(Wout, wout_t, INNER, D_);
  // 3) QKV projection (WMMA + async LDS staging)
  qkv_gemm_kernel<<<(B_ * N_ / 128) * ((3 * INNER) / 64), 256, 0, stream>>>(
      xn_h, wqkv_t, q_h, k_h, v_t);
  // 4) flash attention (WMMA + LDS + streaming softmax)
  attn_kernel<<<B_ * HEADS * (N_ / 16) / 8, 256, 0, stream>>>(q_h, k_h, v_t, m, ao_h);
  // 5) output projection (WMMA + async LDS staging)
  out_gemm_kernel<<<(B_ * N_ / 128) * (D_ / 64), 256, 0, stream>>>(ao_h, wout_t, out);
}
